// PointerGenerator_75265006895153
// MI455X (gfx1250) — compile-verified, matrix operand order checked by reading, not run on hardware
//
#include <hip/hip_runtime.h>
#include <hip/hip_bf16.h>

#define B_ 8
#define S_ 512
#define T_ 256
#define D_ 768
#define H_ 12
#define HID_ 3072
#define V_ 50257
#define O_ 64
#define HD_ 64

typedef __attribute__((ext_vector_type(16))) __bf16 v16bf;
typedef __attribute__((ext_vector_type(8)))  float  v8f;
typedef union { unsigned int u[8]; v16bf v; } FragBF;

// pack two f32 into two bf16 (truncation) with a single v_perm_b32:
// result = { b[31:16], a[31:16] }
__device__ __forceinline__ unsigned int pack_bf16(float a, float b) {
  union { float f; unsigned int u; } x, y;
  x.f = a; y.f = b;
  return __builtin_amdgcn_perm(y.u, x.u, 0x07060302u);
}

// generic shared-memory pointer -> 32-bit LDS byte offset (low 32 address bits)
__device__ __forceinline__ unsigned lds_off(const void* p) {
  return (unsigned)(size_t)p;
}

// ---------------------------------------------------------------------------
// Zero-fill with wide stores
// ---------------------------------------------------------------------------
__global__ void fill_zero4(float4* __restrict__ p, long long n4) {
  long long i = (long long)blockIdx.x * blockDim.x + threadIdx.x;
  if (i < n4) p[i] = make_float4(0.f, 0.f, 0.f, 0.f);
}

// ---------------------------------------------------------------------------
// C[M,N] = act(A[M,K] @ W[N,K]^T + bias[N])   (torch Linear layout)
// 128 threads = 4 waves; block tile 64x64; K step 32; bf16 WMMA, f32 accum.
// Tiles staged to LDS with GLOBAL_LOAD_ASYNC_TO_LDS_B128 (ASYNCcnt pipeline),
// converted to bf16 at fragment-build time (1 v_perm per packed dword).
// Requires M%64==0, N%64==0, K%32==0 (true for all uses here).
// ---------------------------------------------------------------------------
__global__ __launch_bounds__(128) void gemm_bf16w(
    const float* __restrict__ A, const float* __restrict__ W,
    const float* __restrict__ bias, float* __restrict__ C,
    int M, int N, int K, int act) {
  __shared__ float Asf[64][36];   // 64 rows x 32 f32, pitch 144B (16B aligned)
  __shared__ float Bsf[64][36];
  const int tid = threadIdx.x;
  const int l = tid & 31;
  const int w = tid >> 5;
  const int m0 = blockIdx.x * 64;
  const int n0 = blockIdx.y * 64;

  v8f acc[4] = {};

  for (int kk = 0; kk < K; kk += 32) {
    // ---- async stage: 512 x b128 per tile, per-lane LDS destinations ----
#pragma unroll
    for (int i = 0; i < 4; ++i) {
      int idx = i * 128 + tid;
      int rr = idx >> 3, q = idx & 7;            // row 0..63, 16B chunk 0..7
      const float* ga = A + (size_t)(m0 + rr) * K + kk + q * 4;
      unsigned la = lds_off(&Asf[rr][q * 4]);
      asm volatile("global_load_async_to_lds_b128 %0, %1, off"
                   :: "v"(la), "v"(ga) : "memory");
      const float* gb = W + (size_t)(n0 + rr) * K + kk + q * 4;
      unsigned lb = lds_off(&Bsf[rr][q * 4]);
      asm volatile("global_load_async_to_lds_b128 %0, %1, off"
                   :: "v"(lb), "v"(gb) : "memory");
    }
    if (kk + 32 < K) {
      __builtin_prefetch(A + (size_t)(m0 + (tid >> 1)) * K + kk + 32, 0, 0);
      __builtin_prefetch(W + (size_t)(n0 + (tid >> 1)) * K + kk + 32, 0, 0);
    }
    asm volatile("s_wait_asynccnt 0x0" ::: "memory");
    __syncthreads();

    // ---- fragments: f32 pairs from LDS -> packed bf16 ----
    FragBF fa;
#pragma unroll
    for (int p = 0; p < 8; ++p) {
      int k2 = ((p >> 2) << 4) + ((l >> 4) << 3) + ((p & 3) << 1);
      const float2 t2 = *(const float2*)&Asf[w * 16 + (l & 15)][k2];
      fa.u[p] = pack_bf16(t2.x, t2.y);
    }
    FragBF fb[4];
#pragma unroll
    for (int nt = 0; nt < 4; ++nt) {
#pragma unroll
      for (int p = 0; p < 8; ++p) {
        int k2 = ((p >> 2) << 4) + ((l >> 4) << 3) + ((p & 3) << 1);
        const float2 t2 = *(const float2*)&Bsf[nt * 16 + (l & 15)][k2];
        fb[nt].u[p] = pack_bf16(t2.x, t2.y);
      }
    }
    // ---- back-to-back WMMAs (independent accumulators) ----
#pragma unroll
    for (int nt = 0; nt < 4; ++nt) {
      acc[nt] = __builtin_amdgcn_wmma_f32_16x16x32_bf16(
          false, fa.v, false, fb[nt].v, (short)0, acc[nt], false, false);
    }
    __syncthreads();
  }

  // epilogue: bias + optional relu
#pragma unroll
  for (int nt = 0; nt < 4; ++nt) {
    int n = n0 + nt * 16 + (l & 15);
    float bv = bias[n];
#pragma unroll
    for (int r = 0; r < 8; ++r) {
      int m = m0 + w * 16 + r + 8 * (l >> 4);
      float v = acc[nt][r] + bv;
      if (act) v = fmaxf(v, 0.0f);
      C[(size_t)m * N + n] = v;
    }
  }
}

// ---------------------------------------------------------------------------
// Fused attention: per (b, h, 32-row t tile).
//  scores = (Q Kh^T)/8 (+mask) -> softmax -> head-mean atomics -> ctx = P Vh
// ---------------------------------------------------------------------------
__global__ __launch_bounds__(128) void attention_kernel(
    const float* __restrict__ Qb, const float* __restrict__ Kb,
    const float* __restrict__ Vb, const unsigned char* __restrict__ mask,
    float* __restrict__ ctx, float* __restrict__ fs) {
  __shared__ unsigned int Qs[32][33];    // 32 x 64 bf16 (32 dwords/row) + pad
  __shared__ unsigned int KVs[16384];    // K view: [512][32] ; VT view: [64][256]
  __shared__ float Sc[32][516];          // 32 x 512 scores/probs (+pad)

  const int blk = blockIdx.x;
  const int b = blk / (H_ * 8);
  const int rem = blk % (H_ * 8);
  const int h = rem / 8;
  const int t0 = (rem % 8) * 32;
  const int tid = threadIdx.x;
  const int l = tid & 31;
  const int w = tid >> 5;

  // --- stage Q tile (32x64 f32 -> bf16) ---
#pragma unroll
  for (int i = 0; i < 8; ++i) {
    int idx = i * 128 + tid;
    int r = idx >> 5, c = idx & 31;
    const float* qp = Qb + (size_t)(b * T_ + t0 + r) * D_ + h * HD_ + c * 2;
    Qs[r][c] = pack_bf16(qp[0], qp[1]);
  }
  // --- stage K_h (512x64 f32 -> bf16, row-major [s][d]) ---
  for (int i = 0; i < 128; ++i) {
    int idx = i * 128 + tid;
    int s = idx >> 5, c = idx & 31;
    const float* kp = Kb + (size_t)(b * S_ + s) * D_ + h * HD_ + c * 2;
    KVs[idx] = pack_bf16(kp[0], kp[1]);
  }
  __syncthreads();

  // --- scores: wave -> (t half, 256-wide s range) ---
  const int th = w & 1;
  const int sb = (w >> 1) * 256;
  FragBF fa0, fa1;
#pragma unroll
  for (int p = 0; p < 8; ++p) {
    int di = ((p >> 2) << 3) + ((l >> 4) << 2) + (p & 3);
    int row = th * 16 + (l & 15);
    fa0.u[p] = Qs[row][di];
    fa1.u[p] = Qs[row][16 + di];
  }
  for (int st = 0; st < 16; ++st) {
    v8f acc = {};
    FragBF fb0, fb1;
    int n_s = sb + st * 16 + (l & 15);
#pragma unroll
    for (int p = 0; p < 8; ++p) {
      int di = ((p >> 2) << 3) + ((l >> 4) << 2) + (p & 3);
      fb0.u[p] = KVs[n_s * 32 + di];
      fb1.u[p] = KVs[n_s * 32 + 16 + di];
    }
    acc = __builtin_amdgcn_wmma_f32_16x16x32_bf16(
        false, fa0.v, false, fb0.v, (short)0, acc, false, false);
    acc = __builtin_amdgcn_wmma_f32_16x16x32_bf16(
        false, fa1.v, false, fb1.v, (short)0, acc, false, false);
    float mval = mask[b * S_ + n_s] ? -1.0e30f : 0.0f;
#pragma unroll
    for (int r = 0; r < 8; ++r) {
      int m = r + 8 * (l >> 4);
      Sc[th * 16 + m][n_s] = acc[r] * 0.125f + mval;   // 1/sqrt(64)
    }
  }
  __syncthreads();

  // --- restage V_h transposed: VT[d][s-pair] so K(s)-pairs are contiguous ---
  for (int i = 0; i < 128; ++i) {
    int idx = i * 128 + tid;
    int d = idx >> 8, sp = idx & 255;
    const float* vp = Vb + (size_t)(b * S_ + sp * 2) * D_ + h * HD_ + d;
    KVs[idx] = pack_bf16(vp[0], vp[D_]);
  }

  // --- softmax over S (4 threads per row, wave shuffles to combine) ---
  {
    int row = tid >> 2, part = tid & 3;
    float mx = -3.0e38f;
    for (int c = 0; c < 128; ++c) mx = fmaxf(mx, Sc[row][part * 128 + c]);
    mx = fmaxf(mx, __shfl_xor(mx, 1, 4));
    mx = fmaxf(mx, __shfl_xor(mx, 2, 4));
    float sum = 0.f;
    for (int c = 0; c < 128; ++c) {
      float e = __expf(Sc[row][part * 128 + c] - mx);
      Sc[row][part * 128 + c] = e;
      sum += e;
    }
    sum += __shfl_xor(sum, 1, 4);
    sum += __shfl_xor(sum, 2, 4);
    float inv = 1.0f / sum;
    float* fsrow = fs + (size_t)(b * T_ + t0 + row) * S_;
    for (int c = 0; c < 128; ++c) {
      float p = Sc[row][part * 128 + c] * inv;
      Sc[row][part * 128 + c] = p;
      atomicAdd(&fsrow[part * 128 + c], p * (1.0f / (float)H_));
    }
  }
  __syncthreads();

  // --- ctx = P @ V_h : wave -> (t half, 32-wide d half) ---
  const int dhalf = w >> 1;
  v8f c0 = {}, c1 = {};
  for (int s0 = 0; s0 < 512; s0 += 32) {
    FragBF fa;
#pragma unroll
    for (int p = 0; p < 8; ++p) {
      int k2 = ((p >> 2) << 4) + ((l >> 4) << 3) + ((p & 3) << 1);
      int m = th * 16 + (l & 15);
      fa.u[p] = pack_bf16(Sc[m][s0 + k2], Sc[m][s0 + k2 + 1]);
    }
    FragBF fb0, fb1;
    int nd = dhalf * 32 + (l & 15);
#pragma unroll
    for (int p = 0; p < 8; ++p) {
      int di = ((p >> 2) << 3) + ((l >> 4) << 2) + (p & 3);
      fb0.u[p] = KVs[nd * 256 + (s0 >> 1) + di];
      fb1.u[p] = KVs[(nd + 16) * 256 + (s0 >> 1) + di];
    }
    c0 = __builtin_amdgcn_wmma_f32_16x16x32_bf16(
        false, fa.v, false, fb0.v, (short)0, c0, false, false);
    c1 = __builtin_amdgcn_wmma_f32_16x16x32_bf16(
        false, fa.v, false, fb1.v, (short)0, c1, false, false);
  }
#pragma unroll
  for (int r = 0; r < 8; ++r) {
    int m = r + 8 * (l >> 4);
    size_t rowoff = (size_t)(b * T_ + t0 + th * 16 + m) * D_;
    int dcol = h * HD_ + dhalf * 32 + (l & 15);
    ctx[rowoff + dcol]      = c0[r];
    ctx[rowoff + dcol + 16] = c1[r];
  }
}

// ---------------------------------------------------------------------------
// softmax over O=64, in place, one row per thread
// ---------------------------------------------------------------------------
__global__ __launch_bounds__(256) void softmax64_kernel(float* __restrict__ X) {
  int row = blockIdx.x * 256 + threadIdx.x;
  if (row >= B_ * T_) return;
  float* x = X + (size_t)row * O_;
  float mx = x[0];
  for (int i = 1; i < O_; ++i) mx = fmaxf(mx, x[i]);
  float s = 0.f;
  for (int i = 0; i < O_; ++i) { float e = __expf(x[i] - mx); x[i] = e; s += e; }
  float inv = 1.0f / s;
  for (int i = 0; i < O_; ++i) x[i] *= inv;
}

// ---------------------------------------------------------------------------
// copy_probs = sigmoid(ctx_out . copy_w + copy_b), one row per thread
// ---------------------------------------------------------------------------
__global__ __launch_bounds__(256) void copyhead_kernel(
    const float* __restrict__ X, const float* __restrict__ cw,
    const float* __restrict__ cb, float* __restrict__ cp) {
  int row = blockIdx.x * 256 + threadIdx.x;
  if (row >= B_ * T_) return;
  const float* x = X + (size_t)row * D_;
  float s = cb[0];
  for (int d = 0; d < D_; ++d) s += x[d] * cw[d];
  cp[row] = 1.0f / (1.0f + __expf(-s));
}

// ---------------------------------------------------------------------------
// scatter-add into full vocab distribution
// ---------------------------------------------------------------------------
__global__ __launch_bounds__(256) void scatter_kernel(
    const int* __restrict__ src_ids, const int* __restrict__ onto_ids,
    const float* __restrict__ fs, const float* __restrict__ onto_p,
    const float* __restrict__ cp, float* __restrict__ out) {
  int row = blockIdx.x;              // b*T + t
  int b = row / T_;
  float c = cp[row];
  float* orow = out + (size_t)row * V_;
  const float* f = fs + (size_t)row * S_;
  for (int s = threadIdx.x; s < S_; s += 256)
    atomicAdd(&orow[src_ids[b * S_ + s]], f[s] * c);
  const float* op = onto_p + (size_t)row * O_;
  float ic = 1.0f - c;
  for (int o = threadIdx.x; o < O_; o += 256)
    atomicAdd(&orow[onto_ids[o]], op[o] * ic);
}

// ---------------------------------------------------------------------------
extern "C" void kernel_launch(void* const* d_in, const int* in_sizes, int n_in,
                              void* d_out, int out_size, void* d_ws, size_t ws_size,
                              hipStream_t stream) {
  const int*   src_ids  = (const int*)d_in[0];
  const float* enc      = (const float*)d_in[1];
  const unsigned char* mask = (const unsigned char*)d_in[2];  // bool array
  const float* dec      = (const float*)d_in[3];
  const int*   onto_ids = (const int*)d_in[4];
  const float* Wq = (const float*)d_in[5];  const float* bq = (const float*)d_in[6];
  const float* Wk = (const float*)d_in[7];  const float* bk = (const float*)d_in[8];
  const float* Wv = (const float*)d_in[9];  const float* bv = (const float*)d_in[10];
  const float* Wo = (const float*)d_in[11]; const float* bo = (const float*)d_in[12];
  const float* cw = (const float*)d_in[13]; const float* cb = (const float*)d_in[14];
  const float* W1 = (const float*)d_in[15]; const float* b1 = (const float*)d_in[16];
  const float* W2 = (const float*)d_in[17]; const float* b2 = (const float*)d_in[18];

  float* out = (float*)d_out;
  float* ws  = (float*)d_ws;

  // workspace layout (floats)
  float* Hbuf = ws;                                   // 2048*3072
  float* Olog = Hbuf + (size_t)2048 * 3072;           // 2048*64 (logits->probs)
  float* Qb   = Olog + (size_t)2048 * 64;             // 2048*768
  float* Kb   = Qb   + (size_t)2048 * 768;            // 4096*768
  float* Vb   = Kb   + (size_t)4096 * 768;            // 4096*768
  float* Ctx  = Vb   + (size_t)4096 * 768;            // 2048*768
  float* Ctxo = Ctx  + (size_t)2048 * 768;            // 2048*768
  float* Fs   = Ctxo + (size_t)2048 * 768;            // 2048*512
  float* Cp   = Fs   + (size_t)2048 * 512;            // 2048

  // 1) zero outputs / accumulators
  long long n4o = (long long)B_ * T_ * V_ / 4;
  fill_zero4<<<(unsigned)((n4o + 255) / 256), 256, 0, stream>>>((float4*)out, n4o);
  long long n4f = (long long)2048 * 512 / 4;
  fill_zero4<<<(unsigned)((n4f + 255) / 256), 256, 0, stream>>>((float4*)Fs, n4f);

  dim3 blk(128);
  // 2) generator FFN
  gemm_bf16w<<<dim3(2048 / 64, 3072 / 64), blk, 0, stream>>>(dec, W1, b1, Hbuf, 2048, 3072, 768, 1);
  gemm_bf16w<<<dim3(2048 / 64, 64 / 64),   blk, 0, stream>>>(Hbuf, W2, b2, Olog, 2048, 64, 3072, 0);
  softmax64_kernel<<<8, 256, 0, stream>>>(Olog);
  // 3) QKV projections
  gemm_bf16w<<<dim3(2048 / 64, 768 / 64), blk, 0, stream>>>(dec, Wq, bq, Qb, 2048, 768, 768, 0);
  gemm_bf16w<<<dim3(4096 / 64, 768 / 64), blk, 0, stream>>>(enc, Wk, bk, Kb, 4096, 768, 768, 0);
  gemm_bf16w<<<dim3(4096 / 64, 768 / 64), blk, 0, stream>>>(enc, Wv, bv, Vb, 4096, 768, 768, 0);
  // 4) attention (scores + softmax + head-mean + ctx)
  attention_kernel<<<B_ * H_ * (T_ / 32), 128, 0, stream>>>(Qb, Kb, Vb, mask, Ctx, Fs);
  // 5) output projection + copy head
  gemm_bf16w<<<dim3(2048 / 64, 768 / 64), blk, 0, stream>>>(Ctx, Wo, bo, Ctxo, 2048, 768, 768, 0);
  copyhead_kernel<<<8, 256, 0, stream>>>(Ctxo, cw, cb, Cp);
  // 6) scatter-add into vocab distribution
  scatter_kernel<<<B_ * T_, 256, 0, stream>>>(src_ids, onto_ids, Fs, Olog, Cp, out);
}
